// MoDGMMSirenHybridModel_2121713845135
// MI455X (gfx1250) — matrix operand order repaired and checked by reading, compiled, not run on hardware
//
#include <hip/hip_runtime.h>

typedef __attribute__((ext_vector_type(16))) _Float16 v16h;
typedef __attribute__((ext_vector_type(8)))  _Float16 v8h;
typedef __attribute__((ext_vector_type(8)))  float    v8f;
typedef __attribute__((ext_vector_type(2)))  float    v2f;

#define WAVES_PER_BLOCK 8
#define TILE_M 16
#define TWO_PI_F 6.283185307179586f

// Saturating fast tanh: 1 - 2/(exp(2x)+1). exp->inf => 1, exp->0 => -1.
__device__ inline float fast_tanh(float x) {
  return 1.f - 2.f * __builtin_amdgcn_rcpf(__expf(2.f * x) + 1.f);
}

// ---- WMMA fragment helpers (CDNA5 16x16x32 f16, wave32 layouts per ISA 7.12.2) ----
// A-matrix 16x32 f16: lane L holds row m = L&15; its 16 halves are two contiguous
// 8-half runs at K = koff+kb and K = koff+kb+16, kb = (L&16) ? 8 : 0.
__device__ inline v16h lds_load_A(const _Float16* base, int stride, int koff) {
  int lane = threadIdx.x & 31;
  int m  = lane & 15;
  int kb = (lane & 16) ? 8 : 0;
  const _Float16* p = base + m * stride + koff + kb;
  v8h lo = *(const v8h*)(p);        // 16B aligned -> ds_load_b128
  v8h hi = *(const v8h*)(p + 16);
  return __builtin_shufflevector(lo, hi, 0, 1, 2, 3, 4, 5, 6, 7,
                                 8, 9, 10, 11, 12, 13, 14, 15);
}

// B-matrix 32x16 f16: lane L holds column n = L&15; its 16 halves are two contiguous
// 8-half runs at K = koff+kb and K = koff+kb+16, kb = (L&16) ? 8 : 0 (same as A).
// Weights staged row-major W[n][Kpad] f16 => two aligned 16-byte LDS reads of row n.
__device__ inline v16h lds_load_B(const _Float16* base, int kpad, int koff, int noff) {
  int lane = threadIdx.x & 31;
  int n  = (lane & 15) + noff;
  int kb = (lane & 16) ? 8 : 0;
  const _Float16* p = base + n * kpad + koff + kb;
  v8h lo = *(const v8h*)(p);        // 16B aligned -> ds_load_b128
  v8h hi = *(const v8h*)(p + 16);
  return __builtin_shufflevector(lo, hi, 0, 1, 2, 3, 4, 5, 6, 7,
                                 8, 9, 10, 11, 12, 13, 14, 15);
}

__device__ inline v8f wmma_f16(v16h a, v16h b, v8f c) {
  return __builtin_amdgcn_wmma_f32_16x16x32_f16(false, a, false, b, (short)0, c, false, false);
}

// C/D 16x16 f32 fragment: lane = column n, VGPR r = row (m = r + 8*(lane>=16)).
template <int MODE> // 0 = relu, 1 = sin
__device__ inline void act_store(_Float16* dst, int stride, int noff, v8f c, const float* bias) {
  int lane = threadIdx.x & 31;
  int n  = lane & 15;
  int mb = (lane & 16) ? 8 : 0;
  float bv = bias[noff + n];
#pragma unroll
  for (int r = 0; r < 8; ++r) {
    float v = c[r] + bv;
    if (MODE == 0) v = fmaxf(v, 0.f);
    else v = __sinf(v);
    dst[(mb + r) * stride + noff + n] = (_Float16)v;
  }
}

__global__ __launch_bounds__(256) void mod_gmm_siren_kernel(
    const float* __restrict__ coords, const float* __restrict__ fgrid,
    const float* __restrict__ W1, const float* __restrict__ b1,
    const float* __restrict__ W2, const float* __restrict__ b2,
    const float* __restrict__ Ws1, const float* __restrict__ bs1,
    const float* __restrict__ Ws2, const float* __restrict__ bs2,
    const float* __restrict__ Wf, const float* __restrict__ bf,
    const float* __restrict__ Wh, const float* __restrict__ bh,
    float* __restrict__ out, int Npts)
{
  // Weights staged ROW-MAJOR (out x Kpad) in f16 — B fragments read contiguously.
  __shared__ __align__(32) _Float16 sW1[128 * 64];   // W1 (128,34) -> Kpad 64
  __shared__ __align__(32) _Float16 sW2[64 * 128];   // W2 (64,128)
  __shared__ __align__(32) _Float16 sWs2[64 * 128];  // Ws2 (64,128)
  __shared__ __align__(32) _Float16 sWf[128 * 64];   // Wf (128,64)
  __shared__ __align__(32) _Float16 sWh[32 * 64];    // Wh (18,64) -> Npad 32
  __shared__ float sb1[128], sbf[128];
  __shared__ float sb2[64], sbs2[64], sbh[32];
  __shared__ __align__(8) float sTime[256];          // interleaved {30*Ws1[j], 30*bs1[j]}
  __shared__ __align__(32) _Float16 sScratch[WAVES_PER_BLOCK][TILE_M * 128];

  const int tid = threadIdx.x;

  // ---- cooperative weight staging (direct f32 -> f16 copies; W1 padded in K, Wh in N) ----
  for (int i = tid; i < 128 * 64; i += 256) {
    int n = i >> 6, k = i & 63;
    sW1[i] = (_Float16)((k < 34) ? W1[n * 34 + k] : 0.f);
    sWf[i] = (_Float16)Wf[i];
  }
  for (int i = tid; i < 64 * 128; i += 256) {
    sW2[i]  = (_Float16)W2[i];
    sWs2[i] = (_Float16)Ws2[i];
  }
  for (int i = tid; i < 32 * 64; i += 256)
    sWh[i] = (_Float16)((i < 18 * 64) ? Wh[i] : 0.f);
  if (tid < 128) {
    sb1[tid] = b1[tid];
    sbf[tid] = bf[tid];
    sTime[2 * tid]     = 30.f * Ws1[tid];   // fold 30*(t*w+b) into staged constants
    sTime[2 * tid + 1] = 30.f * bs1[tid];
  }
  if (tid < 64) { sb2[tid] = b2[tid]; sbs2[tid] = bs2[tid]; }
  if (tid < 32) { sbh[tid] = (tid < 18) ? bh[tid] : 0.f; }
  __syncthreads();

  const int wave = tid >> 5;
  const int lane = tid & 31;
  const long long base = ((long long)blockIdx.x * WAVES_PER_BLOCK + wave) * TILE_M;
  _Float16* scr = sScratch[wave];

  const int m   = lane & 15;
  const int mb  = (lane & 16) ? 8 : 0;
  const int n16 = lane & 15;
  long long p = base + m;
  if (p >= Npts) p = Npts - 1;   // clamp: keep all lanes live (WMMA needs EXEC all-ones)

  // ---- Phase A: bilinear feature interp -> xy_in tile (16 x 64, K-padded) ----
  {
    float cx = coords[p * 3 + 0], cy = coords[p * 3 + 1];
    float xi = (cx + 1.f) * 0.5f * 63.f;
    float yi = (cy + 1.f) * 0.5f * 63.f;
    int x0 = min(max((int)floorf(xi), 0), 62);
    int y0 = min(max((int)floorf(yi), 0), 62);
    float dx = xi - (float)x0, dy = yi - (float)y0;
    float w00 = (1.f - dx) * (1.f - dy), w01 = (1.f - dx) * dy;
    float w10 = dx * (1.f - dy),         w11 = dx * dy;
    const float* g00 = fgrid + ((y0 * 64 + x0) * 32);
    const float* g01 = fgrid + (((y0 + 1) * 64 + x0) * 32);
    const float* g10 = fgrid + ((y0 * 64 + x0 + 1) * 32);
    const float* g11 = fgrid + (((y0 + 1) * 64 + x0 + 1) * 32);
    int f0 = (lane & 16) ? 16 : 0;   // half-lanes split the 32 features of point m
#pragma unroll
    for (int f = 0; f < 16; f += 2) {
      int fc = f0 + f;
      float s0 = w00 * g00[fc]     + w01 * g01[fc]     + w10 * g10[fc]     + w11 * g11[fc];
      float s1 = w00 * g00[fc + 1] + w01 * g01[fc + 1] + w10 * g10[fc + 1] + w11 * g11[fc + 1];
      union { _Float16 h[2]; unsigned u; } pk;
      pk.h[0] = (_Float16)s0; pk.h[1] = (_Float16)s1;
      *(unsigned*)(scr + m * 128 + 2 + fc) = pk.u;   // col(2+fc) even -> 4B aligned
    }
    if (lane < 16) {
      union { _Float16 h[2]; unsigned u; } pk;
      pk.h[0] = (_Float16)cx; pk.h[1] = (_Float16)cy;
      *(unsigned*)(scr + m * 128) = pk.u;
    }
    // zero-pad K = 34..63 (split 34..49 / 50..63 across lane halves), b32 stores
    int zbase = (lane & 16) ? 50 : 34;
    int zcnt  = (lane & 16) ? 7 : 8;
    for (int z = 0; z < zcnt; ++z)
      *(unsigned*)(scr + m * 128 + zbase + 2 * z) = 0u;
  }

  // ---- Phase B: h = relu(xy_in @ W1^T + b1)  (16x64 @ 64x128) ----
  {
    v16h a0 = lds_load_A(scr, 128, 0);
    v16h a1 = lds_load_A(scr, 128, 32);
#pragma unroll
    for (int nt = 0; nt < 8; ++nt) {
      v8f c = {};
      c = wmma_f16(a0, lds_load_B(sW1, 64, 0, nt * 16), c);
      c = wmma_f16(a1, lds_load_B(sW1, 64, 32, nt * 16), c);
      act_store<0>(scr, 128, nt * 16, c, sb1);
    }
  }

  // ---- Phase C: h_xy = relu(h @ W2^T + b2)  (16x128 @ 128x64), kept in regs ----
  v8f hxy[4];
  {
    v16h a[4];
#pragma unroll
    for (int kt = 0; kt < 4; ++kt) a[kt] = lds_load_A(scr, 128, kt * 32);
#pragma unroll
    for (int nt = 0; nt < 4; ++nt) {
      v8f c = {};
#pragma unroll
      for (int kt = 0; kt < 4; ++kt)
        c = wmma_f16(a[kt], lds_load_B(sW2, 128, kt * 32, nt * 16), c);
      float bv = sb2[nt * 16 + n16];
#pragma unroll
      for (int r = 0; r < 8; ++r) hxy[nt][r] = fmaxf(c[r] + bv, 0.f);
    }
  }

  // ---- Phase D: SIREN time branch ht = sin(t*(30*Ws1) + 30*bs1) -> tile 16x128 ----
  {
    float t = coords[p * 3 + 2];
    int j0 = (lane & 16) ? 64 : 0;
#pragma unroll
    for (int jb = 0; jb < 64; jb += 8) {
      v8h pk;
#pragma unroll
      for (int j = 0; j < 8; ++j) {
        v2f wb = *(const v2f*)(sTime + 2 * (j0 + jb + j));  // ds_load_b64
        pk[j] = (_Float16)__sinf(fmaf(t, wb[0], wb[1]));
      }
      *(v8h*)(scr + m * 128 + j0 + jb) = pk;   // 16B aligned -> ds_store_b128
    }
  }

  // ---- Phase E: h_t = sin(ht @ Ws2^T + bs2) (16x128 @ 128x64) -> tile 16x64 ----
  {
    v16h a[4];
#pragma unroll
    for (int kt = 0; kt < 4; ++kt) a[kt] = lds_load_A(scr, 128, kt * 32);
#pragma unroll
    for (int nt = 0; nt < 4; ++nt) {
      v8f c = {};
#pragma unroll
      for (int kt = 0; kt < 4; ++kt)
        c = wmma_f16(a[kt], lds_load_B(sWs2, 128, kt * 32, nt * 16), c);
      act_store<1>(scr, 128, nt * 16, c, sbs2);
    }
  }

  // ---- Phase F: gamma_beta = tanh(h_t @ Wf^T + bf) (16x64 @ 64x128), in regs ----
  v8f gb[8];
  {
    v16h a0 = lds_load_A(scr, 128, 0);
    v16h a1 = lds_load_A(scr, 128, 32);
#pragma unroll
    for (int nt = 0; nt < 8; ++nt) {
      v8f c = {};
      c = wmma_f16(a0, lds_load_B(sWf, 64, 0, nt * 16), c);
      c = wmma_f16(a1, lds_load_B(sWf, 64, 32, nt * 16), c);
      float bv = sbf[nt * 16 + n16];
#pragma unroll
      for (int r = 0; r < 8; ++r) gb[nt][r] = fast_tanh(c[r] + bv);
    }
  }

  // ---- Phase G: FiLM fuse: (1 + 0.1*gamma)*h_xy + beta -> tile 16x64 ----
#pragma unroll
  for (int nt = 0; nt < 4; ++nt) {
#pragma unroll
    for (int r = 0; r < 8; ++r) {
      float v = (1.f + 0.1f * gb[nt][r]) * hxy[nt][r] + gb[nt + 4][r];
      scr[(mb + r) * 128 + nt * 16 + n16] = (_Float16)v;
    }
  }

  // ---- Phase H: params = fused @ Wh^T + bh (16x64 @ 64x32) -> f32 tile 16x32 ----
  float* pf = (float*)scr;   // reuse wave scratch as 16x32 f32
  {
    v16h a0 = lds_load_A(scr, 128, 0);
    v16h a1 = lds_load_A(scr, 128, 32);
#pragma unroll
    for (int nt = 0; nt < 2; ++nt) {
      v8f c = {};
      c = wmma_f16(a0, lds_load_B(sWh, 64, 0, nt * 16), c);
      c = wmma_f16(a1, lds_load_B(sWh, 64, 32, nt * 16), c);
      float bv = sbh[nt * 16 + n16];
#pragma unroll
      for (int r = 0; r < 8; ++r)
        pf[(mb + r) * 32 + nt * 16 + n16] = c[r] + bv;
    }
  }

  // ---- Phase I: per-point epilogue (softmax/relu/mod/exp/tanh) + output ----
  if (lane < 16) {
    long long q = base + lane;
    if (q < Npts) {
      const float* row = pf + lane * 32;
      float l0 = row[0], l1 = row[6], l2 = row[12];
      float mx = fmaxf(l0, fmaxf(l1, l2));
      float e0 = __expf(l0 - mx), e1 = __expf(l1 - mx), e2 = __expf(l2 - mx);
      float inv = 1.f / (e0 + e1 + e2);
      float w[3] = {e0 * inv, e1 * inv, e2 * inv};
      float* o = out + q * 18;
#pragma unroll
      for (int k = 0; k < 3; ++k) {
        float sp = fmaxf(row[k * 6 + 1], 0.f);
        float an = row[k * 6 + 2];
        an = an - floorf(an / TWO_PI_F) * TWO_PI_F;
        float v3 = __expf(fminf(fmaxf(row[k * 6 + 3], -10.f), 10.f));
        float v4 = __expf(fminf(fmaxf(row[k * 6 + 4], -10.f), 10.f));
        float cr = 0.99f * fast_tanh(row[k * 6 + 5]);
        o[k * 6 + 0] = w[k];
        o[k * 6 + 1] = sp;
        o[k * 6 + 2] = an;
        o[k * 6 + 3] = v3;
        o[k * 6 + 4] = v4;
        o[k * 6 + 5] = cr;
      }
    }
  } else {
    long long q = base + (lane - 16);
    if (q < Npts) {
      float* oc = out + 18LL * Npts;
#pragma unroll
      for (int j = 0; j < 3; ++j) oc[q * 3 + j] = coords[q * 3 + j];
    }
  }
}

extern "C" void kernel_launch(void* const* d_in, const int* in_sizes, int n_in,
                              void* d_out, int out_size, void* d_ws, size_t ws_size,
                              hipStream_t stream) {
  const float* coords = (const float*)d_in[0];
  const float* fgrid  = (const float*)d_in[1];
  const float* W1  = (const float*)d_in[2];
  const float* b1  = (const float*)d_in[3];
  const float* W2  = (const float*)d_in[4];
  const float* b2  = (const float*)d_in[5];
  const float* Ws1 = (const float*)d_in[6];
  const float* bs1 = (const float*)d_in[7];
  const float* Ws2 = (const float*)d_in[8];
  const float* bs2 = (const float*)d_in[9];
  const float* Wf  = (const float*)d_in[10];
  const float* bf  = (const float*)d_in[11];
  const float* Wh  = (const float*)d_in[12];
  const float* bh  = (const float*)d_in[13];
  float* out = (float*)d_out;

  int Npts = in_sizes[0] / 3;
  long long tiles = ((long long)Npts + TILE_M - 1) / TILE_M;
  int blocks = (int)((tiles + WAVES_PER_BLOCK - 1) / WAVES_PER_BLOCK);

  mod_gmm_siren_kernel<<<blocks, 256, 0, stream>>>(
      coords, fgrid, W1, b1, W2, b2, Ws1, bs1, Ws2, bs2, Wf, bf, Wh, bh, out, Npts);
}
